// RegionProposalNetwork_73280732004497
// MI455X (gfx1250) — matrix operand order repaired and checked by reading, not compiled, not run on hardware
//
#include <hip/hip_runtime.h>
#include <hip/hip_bf16.h>
#include <math.h>

// ---------------- problem constants ----------------
#define B_   2
#define CCH  256
#define GH   100
#define GW   100
#define H2   102              // padded height (1 zero row each side)
#define W2   116              // padded width (1 left + 15 right ghost cols)
#define NPIX (GH*GW)          // 10000
#define AANCH 9
#define K3   (CCH*9)          // 2304
#define NSTEP (K3/32)         // 72 K-steps for conv
#define NE   (NPIX*AANCH)     // 90000
#define PRENMS 2000
#define TOPK_  1000
#define SELCAP 2048
#define NMSW   64
#define HEADR  48             // 9 cls + 36 box padded to 48

typedef __attribute__((ext_vector_type(16))) __bf16 v16bf;
typedef __attribute__((ext_vector_type(8)))  __bf16 v8bf;
typedef __attribute__((ext_vector_type(8)))  float  v8f;

// ---------------- weight swizzle: per-lane ISA A-layout, tap-major K ----------------
// wAsw[((mt*72 + step)*32 + lane)*16 + i] ; k = step*32 + kk, kk = 16*hi + i (contiguous)
__global__ void __launch_bounds__(256) swizzle_conv_w_k(const float* __restrict__ conv_w,
                                                        __bf16* __restrict__ wAsw) {
  int i = blockIdx.x * 256 + threadIdx.x;
  if (i >= 16 * NSTEP * 32 * 16) return;
  int i4   = i & 15;
  int lane = (i >> 4) & 31;
  int g    = i >> 9;             // mt*72 + step
  int step = g % NSTEP;
  int mt   = g / NSTEP;
  int lo = lane & 15, hi = lane >> 4;
  int kk = (hi << 4) + i4;                            // contiguous A element->K map: k = 16*hi + i
  int k  = step * 32 + kk;                            // tap-major: k = tap*256 + ic
  int tap = k >> 8, ic = k & 255;
  int kh = tap / 3, kw = tap % 3;
  int oc = mt * 16 + lo;
  wAsw[i] = (__bf16)conv_w[((oc * CCH + ic) * 3 + kh) * 3 + kw];
}

// wHsw[((mt*8 + step)*32 + lane)*16 + i] for fused 48x256 head weights
__global__ void __launch_bounds__(256) swizzle_head_w_k(const float* __restrict__ cls_w,
                                                        const float* __restrict__ box_w,
                                                        __bf16* __restrict__ wHsw) {
  int i = blockIdx.x * 256 + threadIdx.x;
  if (i >= 3 * 8 * 32 * 16) return;
  int i4   = i & 15;
  int lane = (i >> 4) & 31;
  int g    = i >> 9;
  int step = g % 8;
  int mt   = g / 8;
  int lo = lane & 15, hi = lane >> 4;
  int kk = (hi << 4) + i4;
  int k  = step * 32 + kk;
  int row = mt * 16 + lo;
  float v = 0.f;
  if (row < 9)       v = cls_w[row * CCH + k];
  else if (row < 45) v = box_w[(row - 9) * CCH + k];
  wHsw[i] = (__bf16)v;
}

// ---------------- feat: NCHW f32 -> padded pixel-major bf16 [b][y2][x2][c] ----------------
__global__ void __launch_bounds__(256) pack_feat_k(const float* __restrict__ feat,
                                                   __bf16* __restrict__ featT) {
  int i = blockIdx.x * 256 + threadIdx.x;
  if (i >= B_ * H2 * W2 * CCH) return;
  int c  = i & 255;
  int p  = i >> 8;               // b*H2*W2 + y2*W2 + x2
  int x2 = p % W2;
  int y2 = (p / W2) % H2;
  int b  = p / (W2 * H2);
  int y = y2 - 1, x = x2 - 1;
  float v = 0.f;
  if (y >= 0 && y < GH && x >= 0 && x < GW)
    v = feat[((b * CCH + c) * GH + y) * GW + x];
  featT[i] = (__bf16)v;
}

// ---------------- 3x3 conv + bias + relu via WMMA bf16 ----------------
// One wave per 16(oc) x 16(x) tile at fixed y. tile = ((b*100+y)*7+xt)*16 + mt.
__global__ void __launch_bounds__(32) conv3x3_wmma_k(const __bf16* __restrict__ featT,
                                                     const __bf16* __restrict__ wAsw,
                                                     const float*  __restrict__ conv_b,
                                                     __bf16* __restrict__ rpnT) {
  int tile = blockIdx.x;
  int mt = tile & 15;
  int t4 = tile >> 4;
  int xt = t4 % 7;
  int y  = (t4 / 7) % GH;
  int b  = t4 / 700;
  int lane = threadIdx.x;
  int lo = lane & 15, hi = lane >> 4;
  int x = xt * 16 + lo;          // B column -> pixel x (may be ghost >= 100)

  v8f c = {};
  const __bf16* aptr  = wAsw + ((size_t)mt * NSTEP * 32 + lane) * 16;
  const __bf16* fbase = featT + (size_t)b * H2 * W2 * CCH;

  int step = 0;
#pragma unroll
  for (int kh = 0; kh < 3; ++kh) {
#pragma unroll
    for (int kw = 0; kw < 3; ++kw) {
      const __bf16* pB = fbase + ((size_t)(y + kh) * W2 + (x + kw)) * CCH + hi * 16;
#pragma unroll
      for (int ks = 0; ks < CCH; ks += 32) {
        v16bf a  = *(const v16bf*)(aptr + (size_t)step * 512);  // 32B contiguous
        v16bf bm = *(const v16bf*)(pB + ks);                    // 32B contiguous
        c = __builtin_amdgcn_wmma_f32_16x16x32_bf16(false, a, false, bm,
                                                    (short)0, c, false, false);
        ++step;
      }
    }
  }

  if (x < GW) {
    v8bf o;
#pragma unroll
    for (int r = 0; r < 8; ++r) {
      float v = c[r] + conv_b[mt * 16 + hi * 8 + r];
      o[r] = (__bf16)(v > 0.f ? v : 0.f);
    }
    *(v8bf*)(rpnT + ((size_t)b * NPIX + y * GW + x) * CCH + mt * 16 + hi * 8) = o;
  }
}

// ---------------- fused 1x1 heads (48 rows) via WMMA ----------------
__global__ void __launch_bounds__(32) head_wmma_k(const __bf16* __restrict__ rpnT,
                                                  const __bf16* __restrict__ wHsw,
                                                  float* __restrict__ headout) {
  int tile = blockIdx.x;
  int mt = tile % 3;
  int nt = (tile / 3) % 625;
  int b  = tile / 1875;
  int lane = threadIdx.x;
  int lo = lane & 15, hi = lane >> 4;
  int pix = nt * 16 + lo;

  v8f c = {};
  const __bf16* aptr = wHsw + ((size_t)mt * 8 * 32 + lane) * 16;
  const __bf16* pB   = rpnT + ((size_t)b * NPIX + pix) * CCH + hi * 16;
#pragma unroll
  for (int s = 0; s < 8; ++s) {
    v16bf a  = *(const v16bf*)(aptr + (size_t)s * 512);
    v16bf bm = *(const v16bf*)(pB + s * 32);
    c = __builtin_amdgcn_wmma_f32_16x16x32_bf16(false, a, false, bm,
                                                (short)0, c, false, false);
  }
#pragma unroll
  for (int r = 0; r < 8; ++r) {
    int row = mt * 16 + hi * 8 + r;
    headout[((size_t)b * HEADR + row) * NPIX + pix] = c[r];
  }
}

// ---------------- anchor decode + clip + sigmoid ----------------
__global__ void __launch_bounds__(256) decode_k(const float* __restrict__ headout,
                                                const float* __restrict__ cls_b,
                                                const float* __restrict__ box_b,
                                                const int*   __restrict__ shapes,
                                                float* __restrict__ boxes,
                                                float* __restrict__ scores) {
  int j = blockIdx.x * blockDim.x + threadIdx.x;
  if (j >= B_ * NE) return;
  int b = j / NE, e = j % NE;
  int p = e / AANCH, a = e % AANCH;
  int y = p / GW, x = p % GW;
  int ri = a / 3, si = a % 3;
  const float scales[3] = {128.f, 256.f, 512.f};
  const float ratios[3] = {0.5f, 1.f, 2.f};
  float hr = sqrtf(ratios[ri]);
  float wr = 1.f / hr;
  float ws = wr * scales[si], hs = hr * scales[si];
  float bx0 = rintf(-ws * 0.5f), by0 = rintf(-hs * 0.5f);
  float bx1 = rintf( ws * 0.5f), by1 = rintf( hs * 0.5f);
  float ax0 = x * 16.f + bx0, ay0 = y * 16.f + by0;
  float ax1 = x * 16.f + bx1, ay1 = y * 16.f + by1;
  float wa = ax1 - ax0, ha = ay1 - ay0;
  float cxa = ax0 + 0.5f * wa, cya = ay0 + 0.5f * ha;

  const float* ho = headout + (size_t)b * HEADR * NPIX;
  float sraw = ho[a * NPIX + p] + cls_b[a];
  float dx = ho[(9 + a * 4 + 0) * NPIX + p] + box_b[a * 4 + 0];
  float dy = ho[(9 + a * 4 + 1) * NPIX + p] + box_b[a * 4 + 1];
  float dw = ho[(9 + a * 4 + 2) * NPIX + p] + box_b[a * 4 + 2];
  float dh = ho[(9 + a * 4 + 3) * NPIX + p] + box_b[a * 4 + 3];
  const float CLIP = 4.135166556742356f;  // log(1000/16)
  dw = fminf(dw, CLIP); dh = fminf(dh, CLIP);
  float cx = dx * wa + cxa, cy = dy * ha + cya;
  float w = expf(dw) * wa, h = expf(dh) * ha;
  float x0 = cx - 0.5f * w, y0 = cy - 0.5f * h;
  float x1 = cx + 0.5f * w, y1 = cy + 0.5f * h;
  float Wc = (float)shapes[b * 2 + 1], Hc = (float)shapes[b * 2 + 0];
  x0 = fminf(fmaxf(x0, 0.f), Wc); y0 = fminf(fmaxf(y0, 0.f), Hc);
  x1 = fminf(fmaxf(x1, 0.f), Wc); y1 = fminf(fmaxf(y1, 0.f), Hc);
  float* bb = boxes + ((size_t)b * NE + e) * 4;
  bb[0] = x0; bb[1] = y0; bb[2] = x1; bb[3] = y1;
  scores[(size_t)b * NE + e] = 1.f / (1.f + expf(-sraw));
}

// ---------------- top-2000 via 4-pass radix select + bitonic sort ----------------
__global__ void __launch_bounds__(1024) topk_k(const float* __restrict__ scores,
                                               const float* __restrict__ boxes,
                                               float* __restrict__ selbox,
                                               float* __restrict__ selscore,
                                               int*   __restrict__ selvalid) {
  int b = blockIdx.x, tid = threadIdx.x;
  __shared__ unsigned hist[256];
  __shared__ unsigned s_pref;
  __shared__ int s_rem;
  __shared__ int cnt;
  __shared__ float skey[SELCAP];
  __shared__ int   sidx[SELCAP];
  const float* sc = scores + (size_t)b * NE;

  if (tid == 0) { s_pref = 0; s_rem = PRENMS; }
  __syncthreads();

  for (int pass = 0; pass < 4; ++pass) {
    int sh = 24 - pass * 8;
    if (tid < 256) hist[tid] = 0;
    __syncthreads();
    unsigned pref = s_pref;
    for (int e = tid; e < NE; e += 1024) {
      unsigned key = __float_as_uint(sc[e]);   // sigmoid > 0 -> bits ordered
      bool match = (pass == 0) || ((key >> (sh + 8)) == pref);
      if (match) atomicAdd(&hist[(key >> sh) & 255], 1u);
    }
    __syncthreads();
    if (tid == 0) {
      int cum = 0, rem = s_rem; unsigned d = 0;
      for (int dd = 255; dd >= 0; --dd) {
        int h = (int)hist[dd];
        if (cum + h >= rem) { d = (unsigned)dd; s_rem = rem - cum; break; }
        cum += h;
      }
      s_pref = (s_pref << 8) | d;
    }
    __syncthreads();
  }
  unsigned kstar = s_pref;
  if (tid == 0) cnt = 0;
  __syncthreads();

  for (int e = tid; e < NE; e += 1024) {
    unsigned key = __float_as_uint(sc[e]);
    if (key > kstar) {
      int s = atomicAdd(&cnt, 1);
      if (s < SELCAP) { skey[s] = sc[e]; sidx[s] = e; }
    }
  }
  __syncthreads();
  for (int e = tid; e < NE; e += 1024) {
    unsigned key = __float_as_uint(sc[e]);
    if (key == kstar) {
      int s = atomicAdd(&cnt, 1);
      if (s < SELCAP) { skey[s] = sc[e]; sidx[s] = e; }
    }
  }
  __syncthreads();
  int filled = cnt < SELCAP ? cnt : SELCAP;
  for (int s = filled + tid; s < SELCAP; s += 1024) { skey[s] = -1.f; sidx[s] = -1; }
  __syncthreads();

  for (int k2 = 2; k2 <= SELCAP; k2 <<= 1) {
    for (int jj = k2 >> 1; jj > 0; jj >>= 1) {
      for (int t = tid; t < SELCAP; t += 1024) {
        int ixj = t ^ jj;
        if (ixj > t) {
          bool dir = ((t & k2) == 0);
          if ((skey[t] < skey[ixj]) == dir) {
            float tk = skey[t]; skey[t] = skey[ixj]; skey[ixj] = tk;
            int   ti = sidx[t]; sidx[t] = sidx[ixj]; sidx[ixj] = ti;
          }
        }
      }
      __syncthreads();
    }
  }

  for (int s = tid; s < PRENMS; s += 1024) {
    int idx = sidx[s];
    float x0 = 0, y0 = 0, x1 = 0, y1 = 0, sv = -1.f;
    if (idx >= 0) {
      const float* bb = boxes + ((size_t)b * NE + idx) * 4;
      x0 = bb[0]; y0 = bb[1]; x1 = bb[2]; y1 = bb[3]; sv = skey[s];
    }
    float* ob = selbox + ((size_t)b * SELCAP + s) * 4;
    ob[0] = x0; ob[1] = y0; ob[2] = x1; ob[3] = y1;
    selscore[(size_t)b * SELCAP + s] = sv;
    selvalid[(size_t)b * SELCAP + s] = (x1 - x0 >= 16.f && y1 - y0 >= 16.f) ? 1 : 0;
  }
}

// ---------------- parallel IoU bitmask ----------------
__global__ void __launch_bounds__(NMSW) iou_k(const float* __restrict__ selbox,
                                              unsigned* __restrict__ ov) {
  int i = blockIdx.x % PRENMS;
  int b = blockIdx.x / PRENMS;
  int t = threadIdx.x;
  const float* bb = selbox + (size_t)b * SELCAP * 4;
  float ix0 = bb[i * 4 + 0], iy0 = bb[i * 4 + 1];
  float ix1 = bb[i * 4 + 2], iy1 = bb[i * 4 + 3];
  float iarea = (ix1 - ix0) * (iy1 - iy0);
  unsigned m = 0;
  for (int bit = 0; bit < 32; ++bit) {
    int j = t * 32 + bit;
    if (j < PRENMS && j > i) {
      float jx0 = bb[j * 4 + 0], jy0 = bb[j * 4 + 1];
      float jx1 = bb[j * 4 + 2], jy1 = bb[j * 4 + 3];
      float jarea = (jx1 - jx0) * (jy1 - jy0);
      float lx = fmaxf(ix0, jx0), ly = fmaxf(iy0, jy0);
      float rx = fminf(ix1, jx1), ry = fminf(iy1, jy1);
      float w = fmaxf(rx - lx, 0.f), h = fmaxf(ry - ly, 0.f);
      float inter = w * h;
      float iou = inter / (iarea + jarea - inter + 1e-9f);
      if (iou > 0.7f) m |= (1u << bit);
    }
  }
  ov[((size_t)b * PRENMS + i) * NMSW + t] = m;
}

// ---------------- serial greedy suppression ----------------
__global__ void __launch_bounds__(NMSW) nms_serial_k(const unsigned* __restrict__ ov,
                                                     const int* __restrict__ selvalid,
                                                     int* __restrict__ kept) {
  int b = blockIdx.x, t = threadIdx.x;
  __shared__ unsigned supp[NMSW];
  unsigned w = 0;
  for (int bit = 0; bit < 32; ++bit) {
    int j = t * 32 + bit;
    int v = (j < PRENMS) ? selvalid[(size_t)b * SELCAP + j] : 0;
    if (!v) w |= (1u << bit);
  }
  supp[t] = w;
  __syncthreads();
  for (int i = 0; i < PRENMS; ++i) {
    bool alive = ((supp[i >> 5] >> (i & 31)) & 1u) == 0u;
    __syncthreads();
    if (alive) supp[t] |= ov[((size_t)b * PRENMS + i) * NMSW + t];
    __syncthreads();
  }
  for (int bit = 0; bit < 32; ++bit) {
    int j = t * 32 + bit;
    if (j < PRENMS) kept[(size_t)b * SELCAP + j] = ((supp[t] >> bit) & 1u) ? 0 : 1;
  }
}

// ---------------- final top-1000: kept-first stable partition ----------------
__global__ void __launch_bounds__(256) finalize_k(const float* __restrict__ selbox,
                                                  const float* __restrict__ selscore,
                                                  const int* __restrict__ kept,
                                                  float* __restrict__ out) {
  int b = blockIdx.x, tid = threadIdx.x;
  __shared__ int pos[PRENMS];
  if (tid == 0) {
    int pk = 0;
    for (int s = 0; s < PRENMS; ++s) if (kept[(size_t)b * SELCAP + s]) pos[s] = pk++;
    int ps = pk;
    for (int s = 0; s < PRENMS; ++s) if (!kept[(size_t)b * SELCAP + s]) pos[s] = ps++;
  }
  __syncthreads();
  float* oprop  = out;                       // (B,1000,4)
  float* oscore = out + B_ * TOPK_ * 4;      // (B,1000)
  for (int s = tid; s < PRENMS; s += 256) {
    int p = pos[s];
    if (p < TOPK_) {
      const float* bb = selbox + ((size_t)b * SELCAP + s) * 4;
      float* ob = oprop + ((size_t)b * TOPK_ + p) * 4;
      ob[0] = bb[0]; ob[1] = bb[1]; ob[2] = bb[2]; ob[3] = bb[3];
      oscore[(size_t)b * TOPK_ + p] =
          kept[(size_t)b * SELCAP + s] ? selscore[(size_t)b * SELCAP + s] : -1.f;
    }
  }
}

// ---------------- host launcher ----------------
extern "C" void kernel_launch(void* const* d_in, const int* in_sizes, int n_in,
                              void* d_out, int out_size, void* d_ws, size_t ws_size,
                              hipStream_t stream) {
  const float* feat   = (const float*)d_in[1];
  const int*   shapes = (const int*)d_in[2];
  const float* conv_w = (const float*)d_in[3];
  const float* conv_b = (const float*)d_in[4];
  const float* cls_w  = (const float*)d_in[5];
  const float* cls_b  = (const float*)d_in[6];
  const float* box_w  = (const float*)d_in[7];
  const float* box_b  = (const float*)d_in[8];
  float* out = (float*)d_out;

  char* ws = (char*)d_ws;
  size_t off = 0;
  auto alloc = [&](size_t bytes) {
    size_t o = off;
    off = (off + bytes + 255) & ~(size_t)255;
    return o;
  };
  __bf16* wAsw    = (__bf16*)(ws + alloc((size_t)CCH * K3 * 2));
  __bf16* wHsw    = (__bf16*)(ws + alloc((size_t)HEADR * CCH * 2));
  __bf16* featT   = (__bf16*)(ws + alloc((size_t)B_ * H2 * W2 * CCH * 2));
  __bf16* rpnT    = (__bf16*)(ws + alloc((size_t)B_ * NPIX * CCH * 2));
  float*  headout = (float*)(ws + alloc((size_t)B_ * HEADR * NPIX * 4));
  float*  boxes   = (float*)(ws + alloc((size_t)B_ * NE * 4 * 4));
  float*  scores  = (float*)(ws + alloc((size_t)B_ * NE * 4));
  float*  selbox  = (float*)(ws + alloc((size_t)B_ * SELCAP * 4 * 4));
  float*  selscore= (float*)(ws + alloc((size_t)B_ * SELCAP * 4));
  int*    selvalid= (int*)(ws + alloc((size_t)B_ * SELCAP * 4));
  unsigned* ov    = (unsigned*)(ws + alloc((size_t)B_ * PRENMS * NMSW * 4));
  int*    kept    = (int*)(ws + alloc((size_t)B_ * SELCAP * 4));
  (void)ws_size; (void)in_sizes; (void)n_in; (void)out_size;

  int nWA = 16 * NSTEP * 32 * 16;            // 589824
  swizzle_conv_w_k<<<(nWA + 255) / 256, 256, 0, stream>>>(conv_w, wAsw);
  int nWH = 3 * 8 * 32 * 16;                 // 12288
  swizzle_head_w_k<<<(nWH + 255) / 256, 256, 0, stream>>>(cls_w, box_w, wHsw);
  int nF = B_ * H2 * W2 * CCH;               // 6,057,984
  pack_feat_k<<<(nF + 255) / 256, 256, 0, stream>>>(feat, featT);

  conv3x3_wmma_k<<<B_ * GH * 7 * 16, 32, 0, stream>>>(featT, wAsw, conv_b, rpnT);
  head_wmma_k<<<B_ * 625 * 3, 32, 0, stream>>>(rpnT, wHsw, headout);

  decode_k<<<(B_ * NE + 255) / 256, 256, 0, stream>>>(headout, cls_b, box_b, shapes,
                                                      boxes, scores);
  topk_k<<<B_, 1024, 0, stream>>>(scores, boxes, selbox, selscore, selvalid);
  iou_k<<<B_ * PRENMS, NMSW, 0, stream>>>(selbox, ov);
  nms_serial_k<<<B_, NMSW, 0, stream>>>(ov, selvalid, kept);
  finalize_k<<<B_, 256, 0, stream>>>(selbox, selscore, kept, out);
}
